// Int8BaichuanLayer_49460843380756
// MI455X (gfx1250) — 16-bit and 8-bit matrix operand order repaired and checked by reading, compiled, not run on hardware
//
#include <hip/hip_runtime.h>
#include <stdint.h>

#define HID    4096
#define NHEAD  32
#define HDIM   128
#define INTER  11008
#define SEQ    2048
#define NBATCH 2
#define TOK    (NBATCH * SEQ)

typedef __attribute__((ext_vector_type(8)))  int      v8i;
typedef __attribute__((ext_vector_type(8)))  float    v8f;
typedef __attribute__((ext_vector_type(16))) _Float16 v16h;

union AFrag8 { v8i  v; int64_t d[4]; };
union BFrag8 { v8i  v; int4    q[2]; };
union HFrag  { v16h v; int4    q[2]; };

__device__ __forceinline__ void async_cp16(void* lds_dst, const void* gsrc) {
  uint32_t l = (uint32_t)(uintptr_t)lds_dst;
  asm volatile("global_load_async_to_lds_b128 %0, %1, off"
               :: "v"(l), "v"(gsrc) : "memory");
}
template<int N>
__device__ __forceinline__ void wait_async() {
  asm volatile("s_wait_asynccnt %0" :: "n"(N) : "memory");
}

__global__ __launch_bounds__(256) void rmsnorm_q_k(
    const float* __restrict__ x, const float* __restrict__ w,
    int8_t* __restrict__ out)
{
  const int t = blockIdx.x;
  const float* xr = x + (size_t)t * HID;
  __shared__ float red[8];
  __shared__ float stot;

  float ss = 0.f;
  for (int i = threadIdx.x; i < HID; i += 256) { float v = xr[i]; ss += v * v; }
  #pragma unroll
  for (int m = 16; m >= 1; m >>= 1) ss += __shfl_xor(ss, m, 32);
  if ((threadIdx.x & 31) == 0) red[threadIdx.x >> 5] = ss;
  __syncthreads();
  if (threadIdx.x == 0) {
    float tot = 0.f;
    #pragma unroll
    for (int i = 0; i < 8; ++i) tot += red[i];
    stot = rsqrtf(tot * (1.0f / HID) + 1e-6f);
  }
  __syncthreads();
  const float rr = stot;
  for (int i = threadIdx.x; i < HID; i += 256) {
    float v = xr[i] * rr * w[i];
    v = rintf(v);
    v = fminf(fmaxf(v, -128.f), 127.f);
    out[(size_t)t * HID + i] = (int8_t)v;
  }
}

template<int MODE>
__device__ __forceinline__ void issue_stage(
    const int8_t* A, const int8_t* W1, const int8_t* W2,
    int8_t* bA, int8_t* bB, int8_t* bB2,
    int blockM, int blockN, int K, int k0, int tid)
{
  const int ldr = tid >> 2, ldc = (tid & 3) << 4;
  async_cp16(&bA[ldr * 64 + ldc], &A[(size_t)(blockM + ldr) * K + k0 + ldc]);
  #pragma unroll
  for (int i = 0; i < 2; ++i) {
    int idx = tid + (i << 8);
    int r = idx >> 2, c = (idx & 3) << 4;
    async_cp16(&bB[r * 64 + c], &W1[(size_t)(blockN + r) * K + k0 + c]);
    if (MODE == 2)
      async_cp16(&bB2[r * 64 + c], &W2[(size_t)(blockN + r) * K + k0 + c]);
  }
}

template<int MODE>
__global__ __launch_bounds__(256) void gemm_i8_k(
    const int8_t* __restrict__ A, const int8_t* __restrict__ W1,
    const int8_t* __restrict__ W2,
    const float* __restrict__ bias1, const float* __restrict__ sc1,
    const float* __restrict__ bias2, const float* __restrict__ sc2,
    const float* __restrict__ qs,
    const float* __restrict__ addend, float* __restrict__ outf,
    _Float16* __restrict__ qo, _Float16* __restrict__ ko,
    _Float16* __restrict__ vo, int8_t* __restrict__ oi8, int K)
{
  __shared__ int8_t lA[2][64 * 64];
  __shared__ int8_t lB[2][128 * 64];
  __shared__ int8_t lB2[(MODE == 2) ? 2 * 128 * 64 : 16];

  const int tid  = threadIdx.x;
  const int lane = tid & 31, wid = tid >> 5;
  const int g = lane >> 4, ln = lane & 15;
  const int wm = wid >> 1, wn = wid & 1;
  const int blockM = blockIdx.y << 6;
  const int blockN = blockIdx.x << 7;

  const float S1 = *sc1;
  const float S2 = (MODE == 2) ? *sc2 : 0.f;
  const float QS = (MODE == 2) ? *qs  : 0.f;

  v8i acc[4]  = {};
  v8i acc2[4] = {};

  const int nk = K >> 6;
  issue_stage<MODE>(A, W1, W2, lA[0], lB[0], &lB2[0],
                    blockM, blockN, K, 0, tid);

  for (int kk = 0; kk < nk; ++kk) {
    const int cur = kk & 1;
    const bool more = (kk + 1 < nk);
    if (more) {
      issue_stage<MODE>(A, W1, W2, lA[cur ^ 1], lB[cur ^ 1],
                        &lB2[(MODE == 2) ? (cur ^ 1) * 128 * 64 : 0],
                        blockM, blockN, K, (kk + 1) << 6, tid);
      __builtin_prefetch(&W1[(size_t)(blockN + (tid >> 2)) * K +
                             ((kk + 1) << 6)], 0, 0);
      wait_async<(MODE == 2) ? 5 : 3>();
    } else {
      wait_async<0>();
    }
    __syncthreads();

    AFrag8 af;
    const int arow = (wm * 16 + ln) * 64;
    #pragma unroll
    for (int j = 0; j < 4; ++j)
      af.d[j] = *(const int64_t*)&lA[cur][arow + (j >> 1) * 32 + g * 16 + (j & 1) * 8];

    #pragma unroll
    for (int nt = 0; nt < 4; ++nt) {
      const int brow = (wn * 64 + nt * 16 + ln) * 64;
      BFrag8 bf;
      bf.q[0] = *(const int4*)&lB[cur][brow + g * 16];
      bf.q[1] = *(const int4*)&lB[cur][brow + 32 + g * 16];
      acc[nt] = __builtin_amdgcn_wmma_i32_16x16x64_iu8(
          true, af.v, true, bf.v, acc[nt], false, false);
      if (MODE == 2) {
        BFrag8 bf2;
        bf2.q[0] = *(const int4*)&lB2[cur * 128 * 64 + brow + g * 16];
        bf2.q[1] = *(const int4*)&lB2[cur * 128 * 64 + brow + 32 + g * 16];
        acc2[nt] = __builtin_amdgcn_wmma_i32_16x16x64_iu8(
            true, af.v, true, bf2.v, acc2[nt], false, false);
      }
    }
    __syncthreads();
  }

  #pragma unroll
  for (int nt = 0; nt < 4; ++nt) {
    const int n = blockN + wn * 64 + nt * 16 + ln;
    #pragma unroll
    for (int e = 0; e < 8; ++e) {
      const int m = blockM + wm * 16 + g * 8 + e;
      if (MODE == 0) {
        float val = (float)acc[nt][e] * S1 + bias1[n];
        _Float16 hv = (_Float16)val;
        int which = n >> 12;
        int nn = n & (HID - 1);
        int hh = nn >> 7, d = nn & (HDIM - 1);
        int b = m >> 11, s = m & (SEQ - 1);
        size_t o = (((size_t)b * NHEAD + hh) * SEQ + s) * HDIM + d;
        (which == 0 ? qo : which == 1 ? ko : vo)[o] = hv;
      } else if (MODE == 1) {
        size_t o = (size_t)m * HID + n;
        outf[o] = (float)acc[nt][e] * S1 + bias1[n] + addend[o];
      } else {
        float gv = (float)acc[nt][e]  * S1 + bias1[n];
        float uv = (float)acc2[nt][e] * S2 + bias2[n];
        float h  = (gv / (1.f + __expf(-gv))) * uv;
        float qv = rintf(h / QS);
        qv = fminf(fmaxf(qv, -128.f), 127.f);
        oi8[(size_t)m * INTER + n] = (int8_t)qv;
      }
    }
  }
}

__global__ __launch_bounds__(128) void attn_k(
    const _Float16* __restrict__ Q, const _Float16* __restrict__ K,
    const _Float16* __restrict__ V, const float* __restrict__ oqs,
    int8_t* __restrict__ out_i8)
{
  __shared__ _Float16 lQ[64 * 128];
  __shared__ _Float16 lK[64 * 128];
  __shared__ _Float16 lVt[128 * 64];
  __shared__ _Float16 lP[4][16 * 64];

  const int tid  = threadIdx.x;
  const int lane = tid & 31, wid = tid >> 5;
  const int g = lane >> 4, ln = lane & 15;
  const int qb = blockIdx.x << 6;
  const size_t bh = ((size_t)blockIdx.z * NHEAD + blockIdx.y) * (size_t)SEQ * HDIM;
  const float scale = 0.08838834764831845f;
  const float OQS = *oqs;

  for (int i = tid; i < 1024; i += 128) {
    int r = i >> 4, c = (i & 15) << 3;
    *(int4*)&lQ[r * 128 + c] =
        *(const int4*)&Q[bh + (size_t)(qb + r) * HDIM + c];
  }

  v8f o[8] = {};
  float mrun[8], lrun[8];
  #pragma unroll
  for (int e = 0; e < 8; ++e) { mrun[e] = -1e30f; lrun[e] = 0.f; }

  for (int j0 = 0; j0 <= qb; j0 += 64) {
    __syncthreads();
    for (int i = tid; i < 1024; i += 128) {
      int r = i >> 4, c = (i & 15) << 3;
      async_cp16(&lK[r * 128 + c], &K[bh + (size_t)(j0 + r) * HDIM + c]);
    }
    for (int i = tid; i < 1024; i += 128) {
      int r = i >> 4, c = (i & 15) << 3;
      union { int4 q; _Float16 h[8]; } u;
      u.q = *(const int4*)&V[bh + (size_t)(j0 + r) * HDIM + c];
      #pragma unroll
      for (int e = 0; e < 8; ++e) lVt[(c + e) * 64 + r] = u.h[e];
    }
    wait_async<0>();
    __syncthreads();

    v8f s[4] = {};
    const int qrow = (wid * 16 + ln) * 128;
    #pragma unroll
    for (int c = 0; c < 4; ++c) {
      HFrag qf;
      qf.q[0] = *(const int4*)&lQ[qrow + c * 32 + g * 8];
      qf.q[1] = *(const int4*)&lQ[qrow + c * 32 + 16 + g * 8];
      #pragma unroll
      for (int nt = 0; nt < 4; ++nt) {
        const int krow = (nt * 16 + ln) * 128;
        HFrag bf;
        bf.q[0] = *(const int4*)&lK[krow + c * 32 + g * 8];
        bf.q[1] = *(const int4*)&lK[krow + c * 32 + 16 + g * 8];
        s[nt] = __builtin_amdgcn_wmma_f32_16x16x32_f16(
            false, qf.v, false, bf.v, (short)0, s[nt], false, false);
      }
    }

    const int rbase = qb + wid * 16 + g * 8;
    float mnew[8];
    #pragma unroll
    for (int e = 0; e < 8; ++e) mnew[e] = mrun[e];
    #pragma unroll
    for (int nt = 0; nt < 4; ++nt) {
      const int col = j0 + nt * 16 + ln;
      #pragma unroll
      for (int e = 0; e < 8; ++e) {
        float sv = s[nt][e] * scale;
        sv = (col <= rbase + e) ? sv : -65504.0f;
        s[nt][e] = sv;
        mnew[e] = fmaxf(mnew[e], sv);
      }
    }
    float lpart[8];
    #pragma unroll
    for (int e = 0; e < 8; ++e) {
      #pragma unroll
      for (int msk = 1; msk < 16; msk <<= 1)
        mnew[e] = fmaxf(mnew[e], __shfl_xor(mnew[e], msk, 32));
      float corr = __expf(mrun[e] - mnew[e]);
      lrun[e] *= corr;
      mrun[e] = mnew[e];
      #pragma unroll
      for (int ot = 0; ot < 8; ++ot) o[ot][e] *= corr;
      lpart[e] = 0.f;
    }
    #pragma unroll
    for (int nt = 0; nt < 4; ++nt)
      #pragma unroll
      for (int e = 0; e < 8; ++e) {
        float p = __expf(s[nt][e] - mrun[e]);
        lpart[e] += p;
        lP[wid][(g * 8 + e) * 64 + nt * 16 + ln] = (_Float16)p;
      }
    #pragma unroll
    for (int e = 0; e < 8; ++e) {
      #pragma unroll
      for (int msk = 1; msk < 16; msk <<= 1)
        lpart[e] += __shfl_xor(lpart[e], msk, 32);
      lrun[e] += lpart[e];
    }

    #pragma unroll
    for (int ks = 0; ks < 2; ++ks) {
      HFrag pf;
      pf.q[0] = *(const int4*)&lP[wid][ln * 64 + ks * 32 + g * 8];
      pf.q[1] = *(const int4*)&lP[wid][ln * 64 + ks * 32 + 16 + g * 8];
      #pragma unroll
      for (int ot = 0; ot < 8; ++ot) {
        HFrag vf;
        const int vrow = (ot * 16 + ln) * 64;
        vf.q[0] = *(const int4*)&lVt[vrow + ks * 32 + g * 8];
        vf.q[1] = *(const int4*)&lVt[vrow + ks * 32 + 16 + g * 8];
        o[ot] = __builtin_amdgcn_wmma_f32_16x16x32_f16(
            false, pf.v, false, vf.v, (short)0, o[ot], false, false);
      }
    }
  }

  const int row = qb + wid * 16 + g * 8;
  #pragma unroll
  for (int ot = 0; ot < 8; ++ot)
    #pragma unroll
    for (int e = 0; e < 8; ++e) {
      float val = o[ot][e] / lrun[e];
      float qv = rintf(val / OQS);
      qv = fminf(fmaxf(qv, -128.f), 127.f);
      out_i8[((size_t)blockIdx.z * SEQ + row + e) * HID +
             blockIdx.y * HDIM + ot * 16 + ln] = (int8_t)qv;
    }
}

extern "C" void kernel_launch(void* const* d_in, const int* in_sizes, int n_in,
                              void* d_out, int out_size, void* d_ws, size_t ws_size,
                              hipStream_t stream)
{
  (void)in_sizes; (void)n_in; (void)out_size; (void)ws_size;

  const float*  hidden   = (const float*)d_in[0];
  const float*  ln1_w    = (const float*)d_in[2];
  const float*  ln2_w    = (const float*)d_in[3];
  const int8_t* wpack_w  = (const int8_t*)d_in[4];
  const float*  wpack_b  = (const float*)d_in[5];
  const float*  wpack_a  = (const float*)d_in[6];
  const int8_t* oproj_w  = (const int8_t*)d_in[7];
  const float*  oproj_b  = (const float*)d_in[8];
  const float*  oproj_a  = (const float*)d_in[9];
  const float*  oproj_qs = (const float*)d_in[10];
  const int8_t* gate_w   = (const int8_t*)d_in[11];
  const float*  gate_b   = (const float*)d_in[12];
  const float*  gate_a   = (const float*)d_in[13];
  const int8_t* up_w     = (const int8_t*)d_in[14];
  const float*  up_b     = (const float*)d_in[15];
  const float*  up_a     = (const float*)d_in[16];
  const int8_t* down_w   = (const int8_t*)d_in[17];
  const float*  down_b   = (const float*)d_in[18];
  const float*  down_a   = (const float*)d_in[19];
  const float*  down_qs  = (const float*)d_in[20];

  uint8_t* ws = (uint8_t*)d_ws;
  size_t off = 0;
  auto carve = [&](size_t bytes) -> void* {
    void* p = ws + off;
    off += (bytes + 255) & ~(size_t)255;
    return p;
  };
  int8_t*   x_i8  = (int8_t*)  carve((size_t)TOK * HID);
  _Float16* qbuf  = (_Float16*)carve((size_t)TOK * HID * 2);
  _Float16* kbuf  = (_Float16*)carve((size_t)TOK * HID * 2);
  _Float16* vbuf  = (_Float16*)carve((size_t)TOK * HID * 2);
  int8_t*   ao_i8 = (int8_t*)  carve((size_t)TOK * HID);
  float*    resid = (float*)   carve((size_t)TOK * HID * 4);
  int8_t*   x2_i8 = (int8_t*)  carve((size_t)TOK * HID);
  int8_t*   h_i8  = (int8_t*)  carve((size_t)TOK * INTER);

  rmsnorm_q_k<<<TOK, 256, 0, stream>>>(hidden, ln1_w, x_i8);

  gemm_i8_k<0><<<dim3((3 * HID) / 128, TOK / 64), 256, 0, stream>>>(
      x_i8, wpack_w, nullptr, wpack_b, wpack_a, nullptr, nullptr, nullptr,
      nullptr, nullptr, qbuf, kbuf, vbuf, nullptr, HID);

  attn_k<<<dim3(SEQ / 64, NHEAD, NBATCH), 128, 0, stream>>>(
      qbuf, kbuf, vbuf, oproj_qs, ao_i8);

  gemm_i8_k<1><<<dim3(HID / 128, TOK / 64), 256, 0, stream>>>(
      ao_i8, oproj_w, nullptr, oproj_b, oproj_a, nullptr, nullptr, nullptr,
      hidden, resid, nullptr, nullptr, nullptr, nullptr, HID);

  rmsnorm_q_k<<<TOK, 256, 0, stream>>>(resid, ln2_w, x2_i8);

  gemm_i8_k<2><<<dim3(INTER / 128, TOK / 64), 256, 0, stream>>>(
      x2_i8, gate_w, up_w, gate_b, gate_a, up_b, up_a, down_qs,
      nullptr, nullptr, nullptr, nullptr, nullptr, h_i8, HID);

  gemm_i8_k<1><<<dim3(HID / 128, TOK / 64), 256, 0, stream>>>(
      h_i8, down_w, nullptr, down_b, down_a, nullptr, nullptr, nullptr,
      resid, (float*)d_out, nullptr, nullptr, nullptr, nullptr, INTER);
}
